// MSSM_88476326297814
// MI455X (gfx1250) — matrix operand order repaired and checked by reading, not compiled, not run on hardware
//
#include <hip/hip_runtime.h>
#include <hip/hip_bf16.h>
#include <math.h>

// ---------------- constants from the reference ----------------
#define D_MODEL 512
#define D_INNER 1024
#define D_STATE 16
#define D_CONV  4
#define DT_RANK 64
#define BATCH   2
#define SEQ     4096
#define ROWS    (BATCH * SEQ)   // 8192
#define LN_EPS  1e-5f

// ---------------- vector types for WMMA ----------------
typedef __attribute__((ext_vector_type(16))) __bf16 v16bf;
typedef __attribute__((ext_vector_type(8)))  __bf16 v8bf;
typedef __attribute__((ext_vector_type(8)))  float  v8f;
typedef __attribute__((ext_vector_type(4)))  int    v4i;

// ---------------- CDNA5 async global->LDS copy (guarded) ----------------
#if defined(__HIP_DEVICE_COMPILE__) && __has_builtin(__builtin_amdgcn_global_load_async_to_lds_b128)
#define ASYNC_LDS 1
#else
#define ASYNC_LDS 0
#endif

__device__ __forceinline__ void async_copy_b128(const __bf16* gsrc, __bf16* ldst)
{
#if ASYNC_LDS
    __builtin_amdgcn_global_load_async_to_lds_b128(
        (__attribute__((address_space(1))) v4i*)(void*)gsrc,
        (__attribute__((address_space(3))) v4i*)(void*)ldst,
        0, 0);
#else
    *reinterpret_cast<uint4*>(ldst) = *reinterpret_cast<const uint4*>(gsrc);
#endif
}

__device__ __forceinline__ void wait_async_lds()
{
#if ASYNC_LDS
#if __has_builtin(__builtin_amdgcn_s_wait_asynccnt)
    __builtin_amdgcn_s_wait_asynccnt(0);
#else
    asm volatile("s_wait_asynccnt 0x0" ::: "memory");
#endif
#endif
}

// ======================================================================
// bf16 GEMM, C(f32, MxN) = A(bf16, MxK) * B(bf16, KxN)
// Block tile 128x128, BK=32. 256 threads = 8 waves; each wave owns a
// 64x32 region -> 4x2 = 8 v_wmma_f32_16x16x32_bf16 per k-step.
// A tile staged via async global->LDS (CDNA5), B tile staged transposed.
// Optional bias + activation epilogue (act==2 -> softplus).
// Requires M%128==0, K%32==0, N%8==0 (true for all calls; N edge-guarded
// at chunk granularity since every N here is a multiple of 32).
// ======================================================================
#define BM 128
#define BN 128
#define BK 32

__global__ __launch_bounds__(256)
void gemm_bf16_kernel(const __bf16* __restrict__ A, const __bf16* __restrict__ Bw,
                      float* __restrict__ C, int M, int N, int K,
                      const float* __restrict__ bias, int act)
{
    __shared__ __align__(32) __bf16 As[BM * BK];   // [row][k]          8 KB
    __shared__ __align__(32) __bf16 Bs[BN * BK];   // [col][k] (transp) 8 KB

    const int tid  = threadIdx.x;
    const int m0   = blockIdx.y * BM;
    const int n0   = blockIdx.x * BN;
    const int wave = tid >> 5;
    const int lane = tid & 31;
    const int wm   = (wave & 1) * 64;    // 2 waves tile M
    const int wn   = (wave >> 1) * 32;   // 4 waves tile N
    const int lrow = lane & 15;
    const int hi   = lane >> 4;

    v8f acc[4][2] = {};

    for (int k0 = 0; k0 < K; k0 += BK) {
        // ---- prefetch next k-step tiles toward L2 (global_prefetch_b8) ----
        if (k0 + BK < K) {
            __builtin_prefetch(A + (size_t)(m0 + (tid >> 1)) * K + k0 + BK, 0, 1);
            __builtin_prefetch(Bw + (size_t)(k0 + BK + (tid >> 3)) * N + n0 + (tid & 7) * 16, 0, 1);
        }
        __syncthreads();   // previous tile fully consumed

        // ---- A tile: 128x32 bf16 = 512 x 16B chunks, 2 per thread,
        //      async global->LDS copy (no VGPR bounce) ----
        #pragma unroll
        for (int i = 0; i < 2; ++i) {
            int c   = tid + i * 256;          // 0..511
            int row = c >> 2;                 // 4 chunks per row
            int kc  = (c & 3) * 8;
            async_copy_b128(A + (size_t)(m0 + row) * K + k0 + kc, &As[row * BK + kc]);
        }
        // ---- B tile: 32x128 -> transposed LDS [n][k]; 512 chunks, 2/thread.
        //      Chunk-level edge guard (exact: N % 32 == 0, chunks 8-wide). ----
        #pragma unroll
        for (int i = 0; i < 2; ++i) {
            int c  = tid + i * 256;           // 0..511
            int kk = c >> 4;                  // 0..31
            int nc = (c & 15) * 8;            // 0..120
            __bf16 tmp[8];
            if (n0 + nc + 8 <= N) {
                *reinterpret_cast<uint4*>(tmp) =
                    *reinterpret_cast<const uint4*>(Bw + (size_t)(k0 + kk) * N + n0 + nc);
            } else {
                uint4 z = make_uint4(0u, 0u, 0u, 0u);
                *reinterpret_cast<uint4*>(tmp) = z;
            }
            #pragma unroll
            for (int j = 0; j < 8; ++j)
                Bs[(nc + j) * BK + kk] = tmp[j];
        }
        wait_async_lds();   // ASYNCcnt==0 before the barrier (ds stores fenced by compiler)
        __syncthreads();

        // ---- fragments in the 16x16x32 bf16 WMMA register layout ----
        v16bf afr[4], bfr[2];
        #pragma unroll
        for (int mt = 0; mt < 4; ++mt) {
            const __bf16* ap = &As[(wm + mt * 16 + lrow) * BK];
            v8bf lo  = *reinterpret_cast<const v8bf*>(ap + hi * 8);
            v8bf hi8 = *reinterpret_cast<const v8bf*>(ap + hi * 8 + 16);
            afr[mt] = __builtin_shufflevector(lo, hi8,
                0,1,2,3,4,5,6,7,8,9,10,11,12,13,14,15);
        }
        #pragma unroll
        for (int nt = 0; nt < 2; ++nt) {
            const __bf16* bp = &Bs[(wn + nt * 16 + lrow) * BK + hi * 8];
            v8bf lo  = *reinterpret_cast<const v8bf*>(bp);
            v8bf hi8 = *reinterpret_cast<const v8bf*>(bp + 16);
            bfr[nt] = __builtin_shufflevector(lo, hi8,
                0,1,2,3,4,5,6,7,8,9,10,11,12,13,14,15);
        }
        #pragma unroll
        for (int mt = 0; mt < 4; ++mt)
            #pragma unroll
            for (int nt = 0; nt < 2; ++nt)
                acc[mt][nt] = __builtin_amdgcn_wmma_f32_16x16x32_bf16(
                    false, afr[mt], false, bfr[nt],
                    (short)0, acc[mt][nt], false, false);
    }

    // ---- epilogue + store ----
    #pragma unroll
    for (int mt = 0; mt < 4; ++mt) {
        #pragma unroll
        for (int nt = 0; nt < 2; ++nt) {
            int n = n0 + wn + nt * 16 + lrow;
            if (n >= N) continue;
            int mb = m0 + wm + mt * 16 + hi * 8;
            float bv = bias ? bias[n] : 0.0f;
            #pragma unroll
            for (int r = 0; r < 8; ++r) {
                float v = acc[mt][nt][r] + bv;
                if (act == 2) v = (v > 20.0f) ? v : log1pf(__expf(v));  // softplus
                C[(size_t)(mb + r) * N + n] = v;
            }
        }
    }
}

// ======================================================================
// Elementwise converts
// ======================================================================
__global__ void cvt_f32_bf16_kernel(const float* __restrict__ s,
                                    __bf16* __restrict__ d, int n)
{
    int i = blockIdx.x * 256 + threadIdx.x;
    if (i < n) d[i] = (__bf16)s[i];
}

// xdbl (R x 96) first DT_RANK cols -> bf16 (R x 64)
__global__ void cvt_dtrank_bf16_kernel(const float* __restrict__ xdbl,
                                       __bf16* __restrict__ dst)
{
    int i = blockIdx.x * 256 + threadIdx.x;   // R*64
    int r = i >> 6, c = i & 63;
    dst[i] = (__bf16)xdbl[(size_t)r * 96 + c];
}

// ======================================================================
// fa conv (segment-causal, F segments) + fm conv (causal) + biases,
// then LayerNorm over D_INNER. One block per row. Writes bf16 for GEMM2.
// ======================================================================
__global__ __launch_bounds__(256)
void fused_conv_ln_kernel(const float* __restrict__ xproj, const int* __restrict__ Fp,
                          const float* __restrict__ cws, const float* __restrict__ cbs,
                          const float* __restrict__ cwm, const float* __restrict__ cbm,
                          const float* __restrict__ gamma, const float* __restrict__ beta,
                          __bf16* __restrict__ out_bf)
{
    __shared__ float rs[256], rs2[256];
    const int tid = threadIdx.x;
    const int r   = blockIdx.x;          // 0..ROWS-1
    const int t   = r & (SEQ - 1);
    const int F   = *Fp;
    const int Np  = SEQ / F;
    const int segstart = t - (t % Np);
    const size_t rowbase = (size_t)(r - t);   // b*SEQ

    float vals[4];
    float s = 0.f, s2 = 0.f;
    #pragma unroll
    for (int i = 0; i < 4; ++i) {
        int d = tid + i * 256;
        float acc = cbs[d] + cbm[d];
        #pragma unroll
        for (int k = 0; k < D_CONV; ++k) {
            int o = t - (D_CONV - 1) + k;
            if (o >= segstart)
                acc += cws[d * D_CONV + k] * xproj[(rowbase + o) * 3072 + d];
            if (o >= 0)
                acc += cwm[d * D_CONV + k] * xproj[(rowbase + o) * 3072 + D_INNER + d];
        }
        vals[i] = acc; s += acc; s2 += acc * acc;
    }
    rs[tid] = s; rs2[tid] = s2;
    __syncthreads();
    for (int off = 128; off > 0; off >>= 1) {
        if (tid < off) { rs[tid] += rs[tid + off]; rs2[tid] += rs2[tid + off]; }
        __syncthreads();
    }
    const float mu   = rs[0] * (1.0f / D_INNER);
    const float var  = rs2[0] * (1.0f / D_INNER) - mu * mu;
    const float rstd = rsqrtf(var + LN_EPS);
    #pragma unroll
    for (int i = 0; i < 4; ++i) {
        int d = tid + i * 256;
        float fn = (vals[i] - mu) * rstd * gamma[d] + beta[d];
        out_bf[(size_t)r * D_INNER + d] = (__bf16)fn;
    }
}

// ======================================================================
// mamba conv (causal) + silu on xx half of xz; writes f32 (scan input)
// and bf16 (x_proj GEMM input)
// ======================================================================
__global__ void conv_silu_kernel(const float* __restrict__ xz,
                                 const float* __restrict__ mw, const float* __restrict__ mb,
                                 float* __restrict__ xx, __bf16* __restrict__ xx_bf)
{
    int i = blockIdx.x * 256 + threadIdx.x;          // R*1024
    int d = i & (D_INNER - 1);
    int r = i >> 10;
    int t = r & (SEQ - 1);
    size_t rowbase = (size_t)(r - t);
    float acc = mb[d];
    #pragma unroll
    for (int k = 0; k < D_CONV; ++k) {
        int o = t - (D_CONV - 1) + k;
        if (o >= 0) acc += mw[d * D_CONV + k] * xz[(rowbase + o) * (2 * D_INNER) + d];
    }
    float v = acc / (1.0f + __expf(-acc));           // silu
    xx[i] = v;
    xx_bf[i] = (__bf16)v;
}

// ======================================================================
// Selective scan: one thread per (b,d) channel; 16-state in registers.
// B_t / C_t (shared across all d) staged via LDS each step.
// ======================================================================
__global__ __launch_bounds__(256)
void scan_kernel(const float* __restrict__ dtb, const float* __restrict__ u,
                 const float* __restrict__ xdbl, const float* __restrict__ Alog,
                 const float* __restrict__ Dp, float* __restrict__ y)
{
    __shared__ float sBC[2 * D_STATE];
    const int tid = threadIdx.x;
    const int b   = blockIdx.x >> 2;                       // 4 blocks per batch
    const int d   = ((blockIdx.x & 3) << 8) + tid;

    float A[D_STATE], h[D_STATE];
    #pragma unroll
    for (int s = 0; s < D_STATE; ++s) {
        A[s] = -__expf(Alog[d * D_STATE + s]);
        h[s] = 0.0f;
    }
    const float Dd = Dp[d];
    const size_t base = (size_t)b * SEQ;

    for (int t = 0; t < SEQ; ++t) {
        const size_t row = base + t;
        __syncthreads();
        if (tid < 2 * D_STATE) sBC[tid] = xdbl[row * 96 + DT_RANK + tid];
        __syncthreads();
        const float dt = dtb[row * D_INNER + d];
        const float uu = u[row * D_INNER + d];
        const float du = dt * uu;
        float acc = 0.0f;
        #pragma unroll
        for (int s = 0; s < D_STATE; ++s) {
            float dA = __expf(dt * A[s]);
            h[s] = h[s] * dA + du * sBC[s];
            acc += h[s] * sBC[D_STATE + s];
        }
        y[row * D_INNER + d] = acc + uu * Dd;
    }
}

// ======================================================================
// gating kernels
// ======================================================================
__global__ void gate_mul_kernel(const float* __restrict__ y, const float* __restrict__ xz,
                                __bf16* __restrict__ out_bf)
{
    int i = blockIdx.x * 256 + threadIdx.x;          // R*1024
    int d = i & (D_INNER - 1);
    int r = i >> 10;
    float z = xz[(size_t)r * (2 * D_INNER) + D_INNER + d];
    float g = z / (1.0f + __expf(-z));
    out_bf[i] = (__bf16)(y[i] * g);
}

__global__ void final_gate_kernel(const float* __restrict__ ssm, const float* __restrict__ xproj,
                                  __bf16* __restrict__ out_bf)
{
    int i = blockIdx.x * 256 + threadIdx.x;          // R*1024
    int d = i & (D_INNER - 1);
    int r = i >> 10;
    float fg = xproj[(size_t)r * 3072 + 2 * D_INNER + d];
    float g  = fg / (1.0f + __expf(-fg));
    out_bf[i] = (__bf16)(ssm[i] * g);
}

// ======================================================================
// host-side orchestration
// ======================================================================
extern "C" void kernel_launch(void* const* d_in, const int* in_sizes, int n_in,
                              void* d_out, int out_size, void* d_ws, size_t ws_size,
                              hipStream_t stream)
{
    const float* x       = (const float*)d_in[0];
    const int*   Fp      = (const int*)  d_in[1];
    const float* in_w    = (const float*)d_in[2];
    const float* cws     = (const float*)d_in[3];
    const float* cbs     = (const float*)d_in[4];
    const float* cwm     = (const float*)d_in[5];
    const float* cbm     = (const float*)d_in[6];
    const float* gamma   = (const float*)d_in[7];
    const float* beta    = (const float*)d_in[8];
    const float* m_in_w  = (const float*)d_in[9];
    const float* m_cw    = (const float*)d_in[10];
    const float* m_cb    = (const float*)d_in[11];
    const float* m_x_w   = (const float*)d_in[12];
    const float* m_dt_w  = (const float*)d_in[13];
    const float* m_dt_b  = (const float*)d_in[14];
    const float* m_Alog  = (const float*)d_in[15];
    const float* m_D     = (const float*)d_in[16];
    const float* m_out_w = (const float*)d_in[17];
    const float* out_w   = (const float*)d_in[18];
    float* out = (float*)d_out;

    // ---- workspace layout ----
    char* p = (char*)d_ws;
    auto alloc = [&](size_t bytes) {
        char* q = p; p += (bytes + 255) & ~(size_t)255; return q;
    };
    const size_t R = ROWS;
    __bf16* x_bf    = (__bf16*)alloc(R * D_MODEL * 2);
    __bf16* w1_bf   = (__bf16*)alloc((size_t)D_MODEL * 3072 * 2);
    __bf16* w2_bf   = (__bf16*)alloc((size_t)D_INNER * 2048 * 2);
    __bf16* w3_bf   = (__bf16*)alloc((size_t)D_INNER * 96 * 2);
    __bf16* w4_bf   = (__bf16*)alloc((size_t)DT_RANK * D_INNER * 2);
    __bf16* w5_bf   = (__bf16*)alloc((size_t)D_INNER * D_INNER * 2);
    __bf16* w6_bf   = (__bf16*)alloc((size_t)D_INNER * D_MODEL * 2);
    __bf16* act_bf  = (__bf16*)alloc(R * D_INNER * 2);     // reused bf16 activations
    __bf16* xdbl_bf = (__bf16*)alloc(R * DT_RANK * 2);
    float*  xproj   = (float*)alloc(R * 3072 * 4);         // fa_p | fm_p | fg
    float*  buf1    = (float*)alloc(R * D_INNER * 4);      // dt, later ssm_out
    float*  xz      = (float*)alloc(R * 2048 * 4);         // xx_pre | z
    float*  xx      = (float*)alloc(R * D_INNER * 4);      // silu(conv(xx)) f32
    float*  xdbl    = (float*)alloc(R * 96 * 4);
    float*  ybuf    = (float*)alloc(R * D_INNER * 4);

    auto cvt = [&](const float* s, __bf16* d, size_t n) {
        cvt_f32_bf16_kernel<<<(unsigned)((n + 255) / 256), 256, 0, stream>>>(s, d, (int)n);
    };
    auto gemm = [&](const __bf16* A, const __bf16* Bw, float* C,
                    int M, int N, int K, const float* bias, int act) {
        dim3 g((N + BN - 1) / BN, M / BM);
        gemm_bf16_kernel<<<g, 256, 0, stream>>>(A, Bw, C, M, N, K, bias, act);
    };

    // ---- bf16 conversions ----
    cvt(x,       x_bf,  R * D_MODEL);
    cvt(in_w,    w1_bf, (size_t)D_MODEL * 3072);
    cvt(m_in_w,  w2_bf, (size_t)D_INNER * 2048);
    cvt(m_x_w,   w3_bf, (size_t)D_INNER * 96);
    cvt(m_dt_w,  w4_bf, (size_t)DT_RANK * D_INNER);
    cvt(m_out_w, w5_bf, (size_t)D_INNER * D_INNER);
    cvt(out_w,   w6_bf, (size_t)D_INNER * D_MODEL);

    // ---- pipeline ----
    gemm(x_bf, w1_bf, xproj, R, 3072, D_MODEL, nullptr, 0);                 // x @ in_proj_w
    fused_conv_ln_kernel<<<(unsigned)R, 256, 0, stream>>>(
        xproj, Fp, cws, cbs, cwm, cbm, gamma, beta, act_bf);                // conv+conv+LN
    gemm(act_bf, w2_bf, xz, R, 2048, D_INNER, nullptr, 0);                  // @ m_in_proj_w
    conv_silu_kernel<<<(unsigned)(R * D_INNER / 256), 256, 0, stream>>>(
        xz, m_cw, m_cb, xx, act_bf);                                        // conv + silu
    gemm(act_bf, w3_bf, xdbl, R, 96, D_INNER, nullptr, 0);                  // @ m_xproj_w
    cvt_dtrank_bf16_kernel<<<(unsigned)(R * DT_RANK / 256), 256, 0, stream>>>(
        xdbl, xdbl_bf);
    gemm(xdbl_bf, w4_bf, buf1, R, D_INNER, DT_RANK, m_dt_b, 2);             // dt = softplus(.. + b)
    scan_kernel<<<(unsigned)(BATCH * D_INNER / 256), 256, 0, stream>>>(
        buf1, xx, xdbl, m_Alog, m_D, ybuf);                                 // selective scan
    gate_mul_kernel<<<(unsigned)(R * D_INNER / 256), 256, 0, stream>>>(
        ybuf, xz, act_bf);                                                  // y * silu(z)
    gemm(act_bf, w5_bf, buf1, R, D_INNER, D_INNER, nullptr, 0);             // @ m_out_proj_w
    final_gate_kernel<<<(unsigned)(R * D_INNER / 256), 256, 0, stream>>>(
        buf1, xproj, act_bf);                                               // * silu(fg)
    gemm(act_bf, w6_bf, out, R, D_MODEL, D_INNER, nullptr, 0);              // @ out_proj_w
}
